// ProcessVGAE_43722767073851
// MI455X (gfx1250) — matrix operand order repaired and checked by reading, not compiled, not run on hardware
//
// CDNA5 (gfx1250, MI455X) VGAE forward: register-blocked WMMA-f16 GEMMs
// (2x2 tiles per wave -> 2.0 b128 loads per v_wmma) + L2-resident edge scatter.
#include <hip/hip_runtime.h>

typedef _Float16 v16h  __attribute__((ext_vector_type(16)));
typedef _Float16 half8 __attribute__((ext_vector_type(8)));
typedef float    v8f   __attribute__((ext_vector_type(8)));

static constexpr int kN   = 50000;
static constexpr int kE   = 800000;
static constexpr int kC0  = 96;    // node channels
static constexpr int kC1  = 125;
static constexpr int kC2  = 144;
static constexpr int kCMU = 163;
static constexpr int kCE  = 192;   // edge channels
static constexpr int kMTiles = kN / 16;          // 3125 (N divisible by 16)
static constexpr int kMPairs = (kMTiles + 1) / 2; // 1563 (pad 1 tile of rows)
static constexpr int kNPad   = kN + 16;           // A buffers padded w/ zero rows

#define SHUF16(lo, hi) __builtin_shufflevector(lo, hi, 0,1,2,3,4,5,6,7,8,9,10,11,12,13,14,15)

// ---------------------------------------------------------------------------
// WMMA GEMM, 2x2 register blocking: each wave computes a 32x32 C block as
// four 16x16 WMMA tiles, sharing A/B fragments (4 WMMAs per 8 b128 loads).
// A: [kNPad, KPAD] f16 row-major (zero-padded rows/K).
// Bt:[opad,  KPAD] f16 row-major = W^T, opad multiple of 32.
// ---------------------------------------------------------------------------
template <int KPAD>
__global__ __launch_bounds__(256) void gemm_wmma_f16_2x2(
    const _Float16* __restrict__ A, const _Float16* __restrict__ Bt,
    float* __restrict__ C, int outc)
{
  const int lane = threadIdx.x & 31;
  const int wave = threadIdx.x >> 5;
  const int pair = blockIdx.x * 8 + wave;
  if (pair >= kMPairs) return;

  const int row0 = pair * 32;          // first tile rows [row0, row0+16)
  const int nlo  = lane & 15;
  const int hi   = lane >> 4;          // 0: lanes 0-15, 1: lanes 16-31

  // A fragment (16-bit A 16x32 layout): halves 0..7 = K[k+hi*8 ..],
  // halves 8..15 = K[k+16+hi*8 ..], lane holds row m = rowbase + nlo.
  const _Float16* arow0 = A + (size_t)(row0 + nlo) * KPAD;
  const _Float16* arow1 = arow0 + (size_t)16 * KPAD;
  // B fragment: lane holds col n; halves 0..7 = K[k+hi*8 ..], halves 8..15 = K[k+16+hi*8 ..].
  const int col0 = blockIdx.y * 32 + nlo;
  const _Float16* brow0 = Bt + (size_t)col0 * KPAD;
  const _Float16* brow1 = brow0 + (size_t)16 * KPAD;

  v8f acc00 = {}, acc01 = {}, acc10 = {}, acc11 = {};
#pragma unroll
  for (int k = 0; k < KPAD; k += 32) {
    half8 a0l = *(const half8*)(arow0 + k + hi * 8);
    half8 a0h = *(const half8*)(arow0 + k + 16 + hi * 8);
    half8 a1l = *(const half8*)(arow1 + k + hi * 8);
    half8 a1h = *(const half8*)(arow1 + k + 16 + hi * 8);
    half8 b0l = *(const half8*)(brow0 + k + hi * 8);
    half8 b0h = *(const half8*)(brow0 + k + 16 + hi * 8);
    half8 b1l = *(const half8*)(brow1 + k + hi * 8);
    half8 b1h = *(const half8*)(brow1 + k + 16 + hi * 8);
    v16h a0 = SHUF16(a0l, a0h);
    v16h a1 = SHUF16(a1l, a1h);
    v16h b0 = SHUF16(b0l, b0h);
    v16h b1 = SHUF16(b1l, b1h);
    acc00 = __builtin_amdgcn_wmma_f32_16x16x32_f16(false, a0, false, b0, (short)0, acc00, false, false);
    acc01 = __builtin_amdgcn_wmma_f32_16x16x32_f16(false, a0, false, b1, (short)0, acc01, false, false);
    acc10 = __builtin_amdgcn_wmma_f32_16x16x32_f16(false, a1, false, b0, (short)0, acc10, false, false);
    acc11 = __builtin_amdgcn_wmma_f32_16x16x32_f16(false, a1, false, b1, (short)0, acc11, false, false);
  }

  // C/D layout: element j of acc -> row = j + 8*hi, col = nlo (within tile).
  const int c0 = blockIdx.y * 32 + nlo;
  const int c1 = c0 + 16;
#pragma unroll
  for (int j = 0; j < 8; ++j) {
    const int r0 = row0 + hi * 8 + j;        // always < kN (first tile of pair)
    const int r1 = r0 + 16;                  // may spill past N on last pair
    if (c0 < outc) {
      C[(size_t)r0 * outc + c0] = acc00[j];
      if (r1 < kN) C[(size_t)r1 * outc + c0] = acc10[j];
    }
    if (c1 < outc) {
      C[(size_t)r0 * outc + c1] = acc01[j];
      if (r1 < kN) C[(size_t)r1 * outc + c1] = acc11[j];
    }
  }
}

// ---------------------------------------------------------------------------
// Graph-normalization helpers
// ---------------------------------------------------------------------------
__global__ void zero_f32(float* p, int n) {
  int i = blockIdx.x * blockDim.x + threadIdx.x;
  if (i < n) p[i] = 0.0f;
}

__global__ void count_deg(const long long* __restrict__ col, float* __restrict__ deg, int e) {
  int i = blockIdx.x * blockDim.x + threadIdx.x;
  if (i < e) atomicAdd(&deg[(int)col[i]], 1.0f);
}

__global__ void finish_dinv(float* d, int n) {  // in-place: deg -> (deg+1)^-0.5
  int i = blockIdx.x * blockDim.x + threadIdx.x;
  if (i < n) d[i] = rsqrtf(d[i] + 1.0f);
}

__global__ void make_coef(const long long* __restrict__ row, const long long* __restrict__ col,
                          const float* __restrict__ dinv, float* __restrict__ coef, int e) {
  int i = blockIdx.x * blockDim.x + threadIdx.x;
  if (i < e) coef[i] = dinv[(int)row[i]] * dinv[(int)col[i]];
}

// out = hw * dinv^2 (self loop) + bias — also serves as the scatter init
__global__ void init_selfloop(const float* __restrict__ hw, const float* __restrict__ dinv,
                              const float* __restrict__ bias, float* __restrict__ out,
                              int n, int outc) {
  int i = blockIdx.x * blockDim.x + threadIdx.x;
  if (i >= n * outc) return;
  int r = i / outc, j = i - r * outc;
  float d = dinv[r];
  out[i] = hw[i] * d * d + bias[j];
}

// warp-per-edge gather/scatter: out[col] += hw[row] * coef  (L2-resident)
__global__ __launch_bounds__(256) void scatter_edges(
    const float* __restrict__ hw, const float* __restrict__ coef,
    const long long* __restrict__ row, const long long* __restrict__ col,
    float* __restrict__ out, int e, int outc)
{
  int ei = blockIdx.x * 8 + (threadIdx.x >> 5);
  if (ei >= e) return;
  const int lane = threadIdx.x & 31;
  const int r = (int)row[ei], c = (int)col[ei];
  const float cf = coef[ei];
  const float* src = hw + (size_t)r * outc;
  float*       dst = out + (size_t)c * outc;
  for (int j = lane; j < outc; j += 32)
    atomicAdd(dst + j, src[j] * cf);
}

// ---------------------------------------------------------------------------
// Packing / activation kernels (write kNPad rows; rows >= kN are zero)
// ---------------------------------------------------------------------------
// f32[kN,inc] -> f16[kNPad,kpad] (zero row/K padding). op: 0 = copy, 1 = relu
__global__ void pack_f16(const float* __restrict__ src, _Float16* __restrict__ dst,
                         int inc, int kpad, int op) {
  int i = blockIdx.x * blockDim.x + threadIdx.x;
  if (i >= kNPad * kpad) return;
  int r = i / kpad, k = i - r * kpad;
  float v = (r < kN && k < inc) ? src[(size_t)r * inc + k] : 0.0f;
  if (op == 1) v = fmaxf(v, 0.0f);
  dst[i] = (_Float16)v;
}

// reparametrize: dst = f16( mu + eps * exp(ls) / 10 ), zero row/K padding
__global__ void reparam_f16(const float* __restrict__ mu, const float* __restrict__ ls,
                            const float* __restrict__ eps, _Float16* __restrict__ dst,
                            int inc, int kpad) {
  int i = blockIdx.x * blockDim.x + threadIdx.x;
  if (i >= kNPad * kpad) return;
  int r = i / kpad, k = i - r * kpad;
  float v = 0.0f;
  if (r < kN && k < inc) {
    size_t s = (size_t)r * inc + k;
    v = mu[s] + eps[s] * __expf(ls[s]) * 0.1f;
  }
  dst[i] = (_Float16)v;
}

// W[in,out] (f32) -> Wt[opad,kpad] (f16, transposed, zero-padded)
__global__ void pack_wt(const float* __restrict__ W, _Float16* __restrict__ Wt,
                        int inc, int outc, int kpad, int opad) {
  int i = blockIdx.x * blockDim.x + threadIdx.x;
  if (i >= opad * kpad) return;
  int o = i / kpad, k = i - o * kpad;
  Wt[i] = (o < outc && k < inc) ? (_Float16)W[(size_t)k * outc + o] : (_Float16)0.0f;
}

// op: 0 = sigmoid, 1 = relu
__global__ void finalize(const float* __restrict__ src, float* __restrict__ dst,
                         int n, int op) {
  int i = blockIdx.x * blockDim.x + threadIdx.x;
  if (i >= n) return;
  float v = src[i];
  dst[i] = (op == 0) ? (1.0f / (1.0f + __expf(-v))) : fmaxf(v, 0.0f);
}

// ---------------------------------------------------------------------------
// Host orchestration
// ---------------------------------------------------------------------------
static void launch_gemm(const _Float16* A, const _Float16* Bt, float* C,
                        int kpad, int outc, hipStream_t s) {
  dim3 g((kMPairs + 7) / 8, (outc + 31) / 32);
  switch (kpad) {
    case 96:  gemm_wmma_f16_2x2<96> <<<g, 256, 0, s>>>(A, Bt, C, outc); break;
    case 128: gemm_wmma_f16_2x2<128><<<g, 256, 0, s>>>(A, Bt, C, outc); break;
    case 160: gemm_wmma_f16_2x2<160><<<g, 256, 0, s>>>(A, Bt, C, outc); break;
    case 192: gemm_wmma_f16_2x2<192><<<g, 256, 0, s>>>(A, Bt, C, outc); break;
  }
}

extern "C" void kernel_launch(void* const* d_in, const int* in_sizes, int n_in,
                              void* d_out, int out_size, void* d_ws, size_t ws_size,
                              hipStream_t stream) {
  (void)in_sizes; (void)n_in; (void)out_size; (void)ws_size;

  const float*     x     = (const float*)d_in[0];
  const long long* eidx  = (const long long*)d_in[1];
  const long long* erow  = eidx;
  const long long* ecol  = eidx + kE;
  const float*     eps_e = (const float*)d_in[2];
  const float*     eps_n = (const float*)d_in[3];
  const float *w1e=(const float*)d_in[4],  *b1e=(const float*)d_in[5];
  const float *w2e=(const float*)d_in[6],  *b2e=(const float*)d_in[7];
  const float *wme=(const float*)d_in[8],  *bme=(const float*)d_in[9];
  const float *wle=(const float*)d_in[10], *ble=(const float*)d_in[11];
  const float *w4e=(const float*)d_in[12], *b4e=(const float*)d_in[13];
  const float *w1n=(const float*)d_in[14], *b1n=(const float*)d_in[15];
  const float *wmn=(const float*)d_in[16], *bmn=(const float*)d_in[17];
  const float *wln=(const float*)d_in[18], *bln=(const float*)d_in[19];
  const float *w5n=(const float*)d_in[20], *b5n=(const float*)d_in[21];

  // ---- workspace carve (256B aligned) -------------------------------------
  char* ws = (char*)d_ws;
  auto carve = [&](size_t bytes) -> char* {
    char* p = ws; ws += (bytes + 255) & ~(size_t)255; return p;
  };
  float*    dinv = (float*)carve((size_t)kN * 4);
  float*    coef = (float*)carve((size_t)kE * 4);
  float*    hw   = (float*)carve((size_t)kN * kCE * 4);   // GEMM output
  float*    agg  = (float*)carve((size_t)kN * kCE * 4);   // aggregated conv output
  float*    mu   = (float*)carve((size_t)kN * kCMU * 4);
  float*    lsb  = (float*)carve((size_t)kN * kCMU * 4);
  _Float16* xf   = (_Float16*)carve((size_t)kNPad * 96  * 2);
  _Float16* act  = (_Float16*)carve((size_t)kNPad * 192 * 2);
  _Float16* wt1e = (_Float16*)carve((size_t)128 * 96  * 2);
  _Float16* wt2e = (_Float16*)carve((size_t)160 * 128 * 2);
  _Float16* wtme = (_Float16*)carve((size_t)192 * 160 * 2);
  _Float16* wtle = (_Float16*)carve((size_t)192 * 160 * 2);
  _Float16* wt4e = (_Float16*)carve((size_t)192 * 192 * 2);
  _Float16* wt1n = (_Float16*)carve((size_t)96 * 96 * 2);
  _Float16* wtmn = (_Float16*)carve((size_t)96 * 96 * 2);
  _Float16* wtln = (_Float16*)carve((size_t)96 * 96 * 2);
  _Float16* wt5n = (_Float16*)carve((size_t)96 * 96 * 2);

  auto blks = [](int n) { return (n + 255) / 256; };

  // ---- graph normalization ------------------------------------------------
  zero_f32   <<<blks(kN), 256, 0, stream>>>(dinv, kN);
  count_deg  <<<blks(kE), 256, 0, stream>>>(ecol, dinv, kE);
  finish_dinv<<<blks(kN), 256, 0, stream>>>(dinv, kN);
  make_coef  <<<blks(kE), 256, 0, stream>>>(erow, ecol, dinv, coef, kE);

  // ---- weight transpose+pack to f16 (opad = multiple of 32) ---------------
  auto packw = [&](const float* W, _Float16* Wt, int inc, int outc, int kpad, int opad) {
    pack_wt<<<blks(opad * kpad), 256, 0, stream>>>(W, Wt, inc, outc, kpad, opad);
  };
  packw(w1e, wt1e,  96, 125,  96, 128);
  packw(w2e, wt2e, 125, 144, 128, 160);
  packw(wme, wtme, 144, 163, 160, 192);
  packw(wle, wtle, 144, 163, 160, 192);
  packw(w4e, wt4e, 163, 192, 192, 192);
  packw(w1n, wt1n,  96,  96,  96,  96);
  packw(wmn, wtmn,  96,  96,  96,  96);
  packw(wln, wtln,  96,  96,  96,  96);
  packw(w5n, wt5n,  96,  96,  96,  96);

  pack_f16<<<blks(kNPad * 96), 256, 0, stream>>>(x, xf, 96, 96, /*copy*/0);

  // ---- one GCN conv: gemm -> selfloop+bias init -> edge scatter ----------
  auto conv = [&](const _Float16* inA, int kpad, const _Float16* Wt,
                  const float* bias, int outc, float* dst) {
    launch_gemm(inA, Wt, hw, kpad, outc, stream);
    init_selfloop<<<blks(kN * outc), 256, 0, stream>>>(hw, dinv, bias, dst, kN, outc);
    scatter_edges<<<(kE + 7) / 8, 256, 0, stream>>>(hw, coef, erow, ecol, dst, kE, outc);
  };

  // ---- edge branch --------------------------------------------------------
  conv(xf,  96,  wt1e, b1e, kC1, agg);
  pack_f16<<<blks(kNPad * 128), 256, 0, stream>>>(agg, act, kC1, 128, /*relu*/1);
  conv(act, 128, wt2e, b2e, kC2, agg);
  pack_f16<<<blks(kNPad * 160), 256, 0, stream>>>(agg, act, kC2, 160, /*relu*/1);
  conv(act, 160, wtme, bme, kCMU, mu);
  conv(act, 160, wtle, ble, kCMU, lsb);
  reparam_f16<<<blks(kNPad * 192), 256, 0, stream>>>(mu, lsb, eps_e, act, kCMU, 192);
  conv(act, 192, wt4e, b4e, kCE, agg);
  finalize<<<blks(kN * kCE), 256, 0, stream>>>(agg, (float*)d_out, kN * kCE, /*sigmoid*/0);

  // ---- node branch --------------------------------------------------------
  conv(xf, 96, wt1n, b1n, kC0, agg);
  pack_f16<<<blks(kNPad * 96), 256, 0, stream>>>(agg, act, kC0, 96, /*relu*/1);
  conv(act, 96, wtmn, bmn, kC0, mu);
  conv(act, 96, wtln, bln, kC0, lsb);
  reparam_f16<<<blks(kNPad * 96), 256, 0, stream>>>(mu, lsb, eps_n, act, kC0, 96);
  conv(act, 96, wt5n, b5n, kC0, agg);
  finalize<<<blks(kN * kC0), 256, 0, stream>>>(
      agg, (float*)d_out + (size_t)kN * kCE, kN * kC0, /*relu*/1);
}
